// RandomizedGRU_28097676050542
// MI455X (gfx1250) — matrix operand order repaired and checked by reading, not compiled, not run on hardware
//
#include <hip/hip_runtime.h>
#include <hip/hip_bf16.h>

// RandomizedGRU for MI455X (gfx1250, wave32, WMMA f16).
//
//   Kernel 1 (prep):    swizzle weights (f32 -> f16) into WMMA B-fragment layout.
//   Kernel 2 (gemm_px): parallel WMMA GEMM of the h-independent part:
//                       PX[B*T,768] = X[B*T,256] @ [Wz_x | Wr_x | Wtx]  (f16 out)
//   Kernel 3 (scan):    4 workgroups x 512 threads; each owns 16 batch rows and
//                       runs the serial T-loop with interleaved WMMA chains.
//
// Workspace layout (~102 MB):
//   [0)         WxF  f16 256*768   (393216 B)
//   [393216)    WhF  f16 256*768   (393216 B)
//   [786432)    WdF  f16 256*512   (262144 B)
//   [1048576)   PX   f16 65536*768 (100663296 B)

typedef __attribute__((ext_vector_type(16))) _Float16 v16h;
typedef __attribute__((ext_vector_type(8)))  _Float16 v8h;
typedef __attribute__((ext_vector_type(8)))  float    v8f;
typedef __attribute__((ext_vector_type(4)))  float    v4f;

constexpr int Bsz = 64, Tn = 1024, Dn = 256, Hn = 256;
constexpr int N1  = 768;              // [z | r | tilde] column blocks
constexpr float MAX_MEAN = 1000.0f;
constexpr float MAX_LOG_VAR = 30.0f;

__device__ __forceinline__ float sigmoidf_(float v) {
  return 1.0f / (1.0f + __expf(-v));
}
__device__ __forceinline__ float tanhf_(float v) {
  return 2.0f / (1.0f + __expf(-2.0f * v)) - 1.0f;
}

#define WMMA_F16(a, b, c) \
  __builtin_amdgcn_wmma_f32_16x16x32_f16(false, (a), false, (b), (short)0, (c), false, false)

// ---------------------------------------------------------------------------
// Kernel 1: weight swizzle f32 -> f16 B-fragment layout.
// Fragment linear offset o = ((nt*8 + kb)*32 + lane)*16 + i
//   -> k = kb*32 + (i>>3)*16 + (lane>>4)*8 + (i&7) ;  n = nt*16 + (lane&15)
// ---------------------------------------------------------------------------
__device__ __forceinline__ void frag_decode(int o, int& k, int& n) {
  int i    = o & 15;
  int lane = (o >> 4) & 31;
  int kb   = (o >> 9) & 7;
  int nt   = o >> 12;
  k = kb * 32 + ((i >> 3) << 4) + ((lane >> 4) << 3) + (i & 7);
  n = (nt << 4) + (lane & 15);
}

__global__ void prep_weights_kernel(const float* __restrict__ Wz,
                                    const float* __restrict__ Wr,
                                    const float* __restrict__ Wtx,
                                    const float* __restrict__ Wth,
                                    const float* __restrict__ Wd,
                                    _Float16* __restrict__ WxF,
                                    _Float16* __restrict__ WhF,
                                    _Float16* __restrict__ WdF) {
  const int SZ1 = 256 * N1;           // 196608 (WxF, WhF each)
  int idx = blockIdx.x * blockDim.x + threadIdx.x;   // 0 .. 524287
  int k, n;
  if (idx < SZ1) {                    // WxF: x-part rows of Wz/Wr, plus Wtx
    frag_decode(idx, k, n);
    float v;
    if (n < 256)      v = Wz[k * 256 + n];
    else if (n < 512) v = Wr[k * 256 + (n - 256)];
    else              v = Wtx[k * 256 + (n - 512)];
    WxF[idx] = (_Float16)v;
  } else if (idx < 2 * SZ1) {         // WhF: h-part rows of Wz/Wr, plus Wth
    int o = idx - SZ1;
    frag_decode(o, k, n);
    float v;
    if (n < 256)      v = Wz[(256 + k) * 256 + n];
    else if (n < 512) v = Wr[(256 + k) * 256 + (n - 256)];
    else              v = Wth[k * 256 + (n - 512)];
    WhF[o] = (_Float16)v;
  } else {                            // WdF
    int o = idx - 2 * SZ1;
    frag_decode(o, k, n);
    WdF[o] = (_Float16)Wd[k * 512 + n];
  }
}

// ---------------------------------------------------------------------------
// Kernel 2: PX[B*T, 768] = X[B*T, 256] @ WxF   (f16 out, f32 accumulate)
// Two 16x16 output tiles per wave (independent WMMA chains share one A-frag).
// Tile-pairs: (BT/16)=4096 row-tiles x 24 pairs = 98304 waves, 8 waves/block.
// ---------------------------------------------------------------------------
__global__ void gemm_px_kernel(const float* __restrict__ x,
                               const _Float16* __restrict__ WxF,
                               _Float16* __restrict__ PX) {
  int wid  = blockIdx.x * (blockDim.x >> 5) + (threadIdx.x >> 5);
  int lane = threadIdx.x & 31;
  int mt = wid / 24, pt = wid % 24;
  int nt0 = 2 * pt, nt1 = 2 * pt + 1;
  int m = lane & 15, kh = lane >> 4;
  const float* xr = x + (size_t)(mt * 16 + m) * Dn;

  v8f acc0 = {}, acc1 = {};
#pragma unroll
  for (int kb = 0; kb < 8; ++kb) {
    // A fragment: a[i] = X[row][kb*32 + kh*8 + i], a[8+i] = X[row][kb*32+16+kh*8+i]
    const float* ap = xr + kb * 32 + kh * 8;
    v4f f0 = *(const v4f*)(ap);
    v4f f1 = *(const v4f*)(ap + 4);
    v4f f2 = *(const v4f*)(ap + 16);
    v4f f3 = *(const v4f*)(ap + 20);
    v16h a;
#pragma unroll
    for (int i = 0; i < 4; ++i) {
      a[i]      = (_Float16)f0[i];
      a[4 + i]  = (_Float16)f1[i];
      a[8 + i]  = (_Float16)f2[i];
      a[12 + i] = (_Float16)f3[i];
    }
    v16h b0 = *(const v16h*)(WxF + (((nt0 * 8 + kb) * 32 + lane) << 4));
    v16h b1 = *(const v16h*)(WxF + (((nt1 * 8 + kb) * 32 + lane) << 4));
    acc0 = WMMA_F16(a, b0, acc0);     // independent D chains: no hazard NOPs
    acc1 = WMMA_F16(a, b1, acc1);
  }
#pragma unroll
  for (int r = 0; r < 8; ++r) {
    int grow = mt * 16 + kh * 8 + r;          // C layout: VGPR r -> M = r + kh*8
    PX[(size_t)grow * N1 + nt0 * 16 + m] = (_Float16)acc0[r];
    PX[(size_t)grow * N1 + nt1 * 16 + m] = (_Float16)acc1[r];
  }
}

// ---------------------------------------------------------------------------
// Kernel 3: serial scan. blockIdx.x in [0,4): rows 16*bb .. 16*bb+15.
// 512 threads = 16 waves. Each wave interleaves 3 (h-GEMM) / 2 (dist-GEMM)
// independent accumulation chains to avoid WMMA RAW hazard stalls and to
// reuse the A fragment across tiles.
// ---------------------------------------------------------------------------
__global__ void __launch_bounds__(512)
gru_scan_kernel(const float* __restrict__ h0,
                const float* __restrict__ eps,
                const float* __restrict__ bz,  const float* __restrict__ br,
                const float* __restrict__ btx, const float* __restrict__ bth,
                const float* __restrict__ bd,
                const _Float16* __restrict__ WhF,
                const _Float16* __restrict__ WdF,
                const _Float16* __restrict__ PX,
                float* __restrict__ out) {
  __shared__ _Float16 h16[16 * 256];    // carried hidden state (A-matrix source)
  __shared__ _Float16 hb16[16 * 256];   // h_base (A-matrix source for dist GEMM)
  __shared__ float    Pbuf[16 * 768];   // h-GEMM result, then reused for params

  const int tid  = threadIdx.x;
  const int bb   = blockIdx.x;
  const int lane = tid & 31;
  const int w    = tid >> 5;
  const int m    = lane & 15;
  const int kh   = lane >> 4;

  // init h from h0
#pragma unroll
  for (int j = 0; j < 8; ++j) {
    int it = tid + 512 * j;               // 0..4095
    int mm = it >> 8, n = it & 255;
    h16[it] = (_Float16)h0[(bb * 16 + mm) * Hn + n];
  }
  __syncthreads();

  for (int t = 0; t < Tn; ++t) {
    // ---- h-GEMM: Ph[16,768] = h16 @ WhF ; tiles w, w+16, w+32 --------------
    {
      const int nt0 = w, nt1 = w + 16, nt2 = w + 32;
      v8f acc0 = {}, acc1 = {}, acc2 = {};
#pragma unroll
      for (int kb = 0; kb < 8; ++kb) {
        int abase = m * 256 + kb * 32 + kh * 8;
        v8h lo = *(const v8h*)(&h16[abase]);
        v8h hi = *(const v8h*)(&h16[abase + 16]);
        v16h a = __builtin_shufflevector(lo, hi, 0, 1, 2, 3, 4, 5, 6, 7,
                                                 8, 9, 10, 11, 12, 13, 14, 15);
        v16h b0 = *(const v16h*)(WhF + (((nt0 * 8 + kb) * 32 + lane) << 4));
        v16h b1 = *(const v16h*)(WhF + (((nt1 * 8 + kb) * 32 + lane) << 4));
        v16h b2 = *(const v16h*)(WhF + (((nt2 * 8 + kb) * 32 + lane) << 4));
        acc0 = WMMA_F16(a, b0, acc0);
        acc1 = WMMA_F16(a, b1, acc1);
        acc2 = WMMA_F16(a, b2, acc2);
      }
#pragma unroll
      for (int r = 0; r < 8; ++r) {
        int row = (r + kh * 8) * 768;
        Pbuf[row + nt0 * 16 + m] = acc0[r];
        Pbuf[row + nt1 * 16 + m] = acc1[r];
        Pbuf[row + nt2 * 16 + m] = acc2[r];
      }
    }
    __syncthreads();

    // ---- gates / candidate / h_base ---------------------------------------
#pragma unroll
    for (int j = 0; j < 8; ++j) {
      int it = tid + 512 * j;
      int mm = it >> 8, n = it & 255;
      int grow = bb * 16 + mm;
      size_t pbase = ((size_t)grow * Tn + t) * N1;
      float zx = (float)PX[pbase + n];
      float rx = (float)PX[pbase + 256 + n];
      float tx = (float)PX[pbase + 512 + n];
      float zh = Pbuf[mm * 768 + n];
      float rh = Pbuf[mm * 768 + 256 + n];
      float th = Pbuf[mm * 768 + 512 + n];
      float zv = sigmoidf_(zx + zh + bz[n]);
      float rv = sigmoidf_(rx + rh + br[n]);
      float htil = tanhf_(tx + btx[n] + rv * (th + bth[n]));
      float hv = (float)h16[mm * 256 + n];
      float hb = zv * hv + (1.0f - zv) * htil;
      hb16[mm * 256 + n] = (_Float16)hb;
    }
    __syncthreads();

    // ---- dist-GEMM: params[16,512] = hb16 @ WdF ; tiles w, w+16 -----------
    {
      const int nt0 = w, nt1 = w + 16;
      v8f acc0 = {}, acc1 = {};
#pragma unroll
      for (int kb = 0; kb < 8; ++kb) {
        int abase = m * 256 + kb * 32 + kh * 8;
        v8h lo = *(const v8h*)(&hb16[abase]);
        v8h hi = *(const v8h*)(&hb16[abase + 16]);
        v16h a = __builtin_shufflevector(lo, hi, 0, 1, 2, 3, 4, 5, 6, 7,
                                                 8, 9, 10, 11, 12, 13, 14, 15);
        v16h b0 = *(const v16h*)(WdF + (((nt0 * 8 + kb) * 32 + lane) << 4));
        v16h b1 = *(const v16h*)(WdF + (((nt1 * 8 + kb) * 32 + lane) << 4));
        acc0 = WMMA_F16(a, b0, acc0);
        acc1 = WMMA_F16(a, b1, acc1);
      }
#pragma unroll
      for (int r = 0; r < 8; ++r) {
        int row = (r + kh * 8) * 512;
        Pbuf[row + nt0 * 16 + m] = acc0[r];
        Pbuf[row + nt1 * 16 + m] = acc1[r];
      }
    }
    __syncthreads();

    // ---- clip / exp / rsample / emit / carry ------------------------------
#pragma unroll
    for (int j = 0; j < 8; ++j) {
      int it = tid + 512 * j;
      int mm = it >> 8, n = it & 255;
      int grow = bb * 16 + mm;
      float mean = Pbuf[mm * 512 + n] + bd[n];
      mean = fminf(fmaxf(mean, -MAX_MEAN), MAX_MEAN);
      float lv = Pbuf[mm * 512 + 256 + n] + bd[256 + n];
      lv = fminf(fmaxf(lv, -MAX_LOG_VAR), MAX_LOG_VAR);
      float sd = __expf(0.5f * lv);
      float e  = eps[((size_t)grow * Tn + t) * Hn + n];
      float s  = mean + sd * e;
      out[((size_t)grow * Tn + t) * Hn + n] = s;   // h_ts
      h16[mm * 256 + n] = (_Float16)s;             // carry
      if (t == Tn - 1)
        out[(size_t)Bsz * Tn * Hn + (size_t)grow * Hn + n] = s;  // h_ts[:, -1:, :]
    }
    __syncthreads();
  }
}

// ---------------------------------------------------------------------------
extern "C" void kernel_launch(void* const* d_in, const int* in_sizes, int n_in,
                              void* d_out, int out_size, void* d_ws, size_t ws_size,
                              hipStream_t stream) {
  const float* x   = (const float*)d_in[0];
  const float* h0  = (const float*)d_in[1];
  const float* eps = (const float*)d_in[2];
  const float* Wz  = (const float*)d_in[3];
  const float* bz  = (const float*)d_in[4];
  const float* Wr  = (const float*)d_in[5];
  const float* br  = (const float*)d_in[6];
  const float* Wtx = (const float*)d_in[7];
  const float* btx = (const float*)d_in[8];
  const float* Wth = (const float*)d_in[9];
  const float* bth = (const float*)d_in[10];
  const float* Wd  = (const float*)d_in[11];
  const float* bd  = (const float*)d_in[12];
  float* out = (float*)d_out;

  char* ws = (char*)d_ws;
  _Float16* WxF = (_Float16*)(ws);
  _Float16* WhF = (_Float16*)(ws + 393216);
  _Float16* WdF = (_Float16*)(ws + 786432);
  _Float16* PX  = (_Float16*)(ws + 1048576);

  // 1) swizzle weights into B-fragment layout (524288 elements)
  prep_weights_kernel<<<2048, 256, 0, stream>>>(Wz, Wr, Wtx, Wth, Wd,
                                                WxF, WhF, WdF);
  // 2) parallel x-side GEMM: 98304 tile-pairs, 8 waves/block
  gemm_px_kernel<<<12288, 256, 0, stream>>>(x, WxF, PX);
  // 3) serial scan: 4 workgroups x 512 threads (16 batch rows each)
  gru_scan_kernel<<<4, 512, 0, stream>>>(h0, eps, bz, br, btx, bth, bd,
                                         WhF, WdF, PX, out);
}
